// SlidingAttention_54408645706245
// MI455X (gfx1250) — matrix operand order repaired and checked by reading, not compiled, not run on hardware
//
#include <hip/hip_runtime.h>

// Problem constants (match reference)
#define BB 8
#define SS 4096
#define EE 1024
#define HH 8
#define DD 128
#define NWIN 32      // SS / 128 windows
#define KW 384       // 3 * window

typedef __attribute__((ext_vector_type(16))) _Float16 v16h;
typedef __attribute__((ext_vector_type(8)))  _Float16 v8h;
typedef __attribute__((ext_vector_type(8)))  float    v8f;
typedef __attribute__((ext_vector_type(4)))  float    v4f;

// ---------------- WMMA fragment helpers (wave32, 16x16x32 f16) ----------------
// A (16x32, MxK): lanes 0-15 -> M=lane, halfs[0:16]=K0..15
//                 lanes 16-31 -> M=lane-16, halfs[0:16]=K16..31
__device__ __forceinline__ v16h frag_a_f16(const _Float16* base, int ld, int lane) {
  const int r = lane & 15, hs = lane >> 4;
  const _Float16* p = base + (size_t)r * ld + hs * 16;
  v8h lo = *(const v8h*)p;
  v8h hi = *(const v8h*)(p + 8);
  return __builtin_shufflevector(lo, hi, 0,1,2,3,4,5,6,7,8,9,10,11,12,13,14,15);
}

__device__ __forceinline__ v16h frag_a_f32(const float* base, int ld, int lane) {
  const int r = lane & 15, hs = lane >> 4;
  const float* q = base + (size_t)r * ld + hs * 16;
  v4f a0 = *(const v4f*)(q);
  v4f a1 = *(const v4f*)(q + 4);
  v4f b0 = *(const v4f*)(q + 8);
  v4f b1 = *(const v4f*)(q + 12);
  v16h f;
#pragma unroll
  for (int i = 0; i < 4; ++i) {
    f[i]      = (_Float16)a0[i];
    f[4 + i]  = (_Float16)a1[i];
    f[8 + i]  = (_Float16)b0[i];
    f[12 + i] = (_Float16)b1[i];
  }
  return f;
}

// B (32x16, KxN) loaded from an N-major source Bt[n][k]:
// lanes 0-15 -> N=lane, K 0..15 contiguous; lanes 16-31 -> N=lane-16, K 16..31
__device__ __forceinline__ v16h frag_b_f16(const _Float16* base, int ld, int lane) {
  const int n = lane & 15, ks = lane >> 4;
  const _Float16* p = base + (size_t)n * ld + ks * 16;
  v8h lo = *(const v8h*)p;
  v8h hi = *(const v8h*)(p + 8);
  return __builtin_shufflevector(lo, hi, 0,1,2,3,4,5,6,7,8,9,10,11,12,13,14,15);
}

__device__ __forceinline__ v16h frag_b_f32(const float* base, int ld, int lane) {
  const int n = lane & 15, ks = lane >> 4;
  const float* q = base + (size_t)n * ld + ks * 16;
  v4f a0 = *(const v4f*)(q);
  v4f a1 = *(const v4f*)(q + 4);
  v4f a2 = *(const v4f*)(q + 8);
  v4f a3 = *(const v4f*)(q + 12);
  v16h f;
#pragma unroll
  for (int i = 0; i < 4; ++i) {
    f[i]      = (_Float16)a0[i];
    f[4 + i]  = (_Float16)a1[i];
    f[8 + i]  = (_Float16)a2[i];
    f[12 + i] = (_Float16)a3[i];
  }
  return f;
}

// C/D (16x16 f32): lanes 0-15: VGPR r -> M=r, N=lane; lanes 16-31: M=8+r, N=lane-16
__device__ __forceinline__ void store_c_f16(_Float16* base, int ld, int lane, v8f c) {
  const int col = lane & 15, rh = lane >> 4;
#pragma unroll
  for (int r = 0; r < 8; ++r) base[(size_t)(r + rh * 8) * ld + col] = (_Float16)c[r];
}

__device__ __forceinline__ v8f wmma16(v16h a, v16h b, v8f c) {
  return __builtin_amdgcn_wmma_f32_16x16x32_f16(false, a, false, b, (short)0, c, false, false);
}

// ---------------- Kernels ----------------

__global__ void cvt_f32_f16(const float* __restrict__ in, _Float16* __restrict__ out, int n) {
  int i = blockIdx.x * blockDim.x + threadIdx.x;
  if (i < n) out[i] = (_Float16)in[i];
}

// Row-major projection: Out[b,h,s,o] = sum_d X[b,s,h*D+d] * W[o,d]   (Q and K)
// grid = B*H*(S/64), block = 128 (4 waves, 16 s-rows each)
__global__ __launch_bounds__(128) void proj_rowmajor(
    const float* __restrict__ X, const _Float16* __restrict__ Wh, _Float16* __restrict__ Out) {
  const int nsb = SS / 64;
  int sblk = blockIdx.x % nsb;
  int bh   = blockIdx.x / nsb;
  int h    = bh % HH, b = bh / HH;
  int wave = threadIdx.x >> 5, lane = threadIdx.x & 31;
  int m0 = sblk * 64 + wave * 16;

  const float* Xb = X + ((size_t)b * SS + m0) * EE + h * DD;     // A: rows s, ld EE
  _Float16* Ob = Out + ((size_t)bh * SS + m0) * DD;              // out rows s, ld DD

  v16h a[4];
#pragma unroll
  for (int kt = 0; kt < 4; ++kt) a[kt] = frag_a_f32(Xb + kt * 32, EE, lane);

#pragma unroll
  for (int nt = 0; nt < 8; ++nt) {
    v8f c = {};
#pragma unroll
    for (int kt = 0; kt < 4; ++kt) {
      v16h bf = frag_b_f16(Wh + (size_t)(nt * 16) * DD + kt * 32, DD, lane);
      c = wmma16(a[kt], bf, c);
    }
    store_c_f16(Ob + nt * 16, DD, lane, c);
  }
}

// Transposed V projection: Vt[b,h,o,s] = sum_d W[o,d] * X[b,s,h*D+d]
// A = Wv (row-major), B = X^T (columns s contiguous in d). grid = B*H*(S/64), block 128
__global__ __launch_bounds__(128) void proj_v_t(
    const float* __restrict__ X, const _Float16* __restrict__ Wh, _Float16* __restrict__ Vt) {
  const int nsb = SS / 64;
  int sblk = blockIdx.x % nsb;
  int bh   = blockIdx.x / nsb;
  int h    = bh % HH, b = bh / HH;
  int wave = threadIdx.x >> 5, lane = threadIdx.x & 31;

  const float* Xb = X + ((size_t)b * SS + sblk * 64) * EE + h * DD;  // Bt rows s, ld EE
  _Float16* Vb = Vt + (size_t)bh * DD * SS;

  for (int ot = wave; ot < 8; ot += 4) {
    v16h a[4];
#pragma unroll
    for (int kt = 0; kt < 4; ++kt)
      a[kt] = frag_a_f16(Wh + (size_t)(ot * 16) * DD + kt * 32, DD, lane);
#pragma unroll
    for (int st = 0; st < 4; ++st) {
      v8f c = {};
#pragma unroll
      for (int kt = 0; kt < 4; ++kt) {
        v16h bf = frag_b_f32(Xb + (size_t)(st * 16) * EE + kt * 32, EE, lane);
        c = wmma16(a[kt], bf, c);
      }
      store_c_f16(Vb + (size_t)(ot * 16) * SS + sblk * 64 + st * 16, SS, lane, c);
    }
  }
}

// Sliding-window attention, one (b,h,window) per 128-thread block.
// LDS: sim f32 [128][388] then P f16 [128][392]
#define SIMLD 388
#define PLD   392
#define ATTN_SMEM (128 * SIMLD * 4 + 128 * PLD * 2)

__global__ __launch_bounds__(128) void attn_window(
    const _Float16* __restrict__ Qh, const _Float16* __restrict__ Kh,
    const _Float16* __restrict__ Vt, _Float16* __restrict__ AO) {
  extern __shared__ char smem[];
  float*    sim = (float*)smem;
  _Float16* P   = (_Float16*)(smem + 128 * SIMLD * 4);

  int win = blockIdx.x % NWIN;
  int bh  = blockIdx.x / NWIN;
  int h   = bh % HH, b = bh / HH;
  int wave = threadIdx.x >> 5, lane = threadIdx.x & 31;

  const _Float16* Qb = Qh + ((size_t)bh * SS + win * 128) * DD;  // q rows, ld DD
  const _Float16* Kb = Kh + (size_t)bh * SS * DD;                // key rows, ld DD
  const _Float16* Vb = Vt + (size_t)bh * DD * SS;                // Bt rows d, ld SS
  const int s_base = (win - 1) * 128;                            // kk -> s = s_base + kk
  const float scale = 0.08838834764831845f;                      // 1/sqrt(128)

  // ---- Phase 1: sim[128][384] = scale * Q Kg^T ----
  // m-strips outer so Q A-fragments are loaded once and reused across 24 n-tiles.
  for (int mt = wave; mt < 8; mt += 4) {
    v16h a[4];
#pragma unroll
    for (int kt = 0; kt < 4; ++kt)
      a[kt] = frag_a_f16(Qb + (size_t)(mt * 16) * DD + kt * 32, DD, lane);

    for (int nt = 0; nt < 24; ++nt) {
      v8f c = {};
#pragma unroll
      for (int kt = 0; kt < 4; ++kt) {
        // B-fragment from K rows with boundary clamp (masked later)
        const int n = lane & 15, ks = lane >> 4;
        int srow = s_base + nt * 16 + n;
        srow = srow < 0 ? 0 : (srow > SS - 1 ? SS - 1 : srow);
        const _Float16* p = Kb + (size_t)srow * DD + kt * 32 + ks * 16;
        v8h lo = *(const v8h*)p;
        v8h hi = *(const v8h*)(p + 8);
        v16h bf = __builtin_shufflevector(lo, hi, 0,1,2,3,4,5,6,7,8,9,10,11,12,13,14,15);
        c = wmma16(a[kt], bf, c);
      }
      const int col = lane & 15, rh = lane >> 4;
      float* srow = sim + (size_t)(mt * 16 + rh * 8) * SIMLD + nt * 16 + col;
#pragma unroll
      for (int r = 0; r < 8; ++r) srow[(size_t)r * SIMLD] = c[r] * scale;
    }
  }
  __syncthreads();

  // ---- Phase 2: masked softmax, one row per thread ----
  {
    int q = threadIdx.x;            // 0..127
    float* row = sim + (size_t)q * SIMLD;
    const int lo = (win == 0) ? 128 : 0;
    const int hi = (win == NWIN - 1) ? 256 : 384;
    float m = -1e30f;
    for (int k = lo; k < hi; ++k) m = fmaxf(m, row[k]);
    float sum = 0.f;
    for (int k = 0; k < 384; ++k) {
      float e = (k >= lo && k < hi) ? __expf(row[k] - m) : 0.f;
      sum += e;
      row[k] = e;
    }
    float inv = 1.f / sum;
    _Float16* prow = P + (size_t)q * PLD;
    for (int k = 0; k < 384; ++k) prow[k] = (_Float16)(row[k] * inv);
  }
  __syncthreads();

  // ---- Phase 3: out = P[128x384] * Vg[384x128] ----
  // m-strips outer; all 12 K-step A-fragments of the strip cached in registers,
  // inner loop is pure V-fragment load + WMMA.
  for (int mt = wave; mt < 8; mt += 4) {
    v16h a[12];
#pragma unroll
    for (int kt = 0; kt < 12; ++kt)
      a[kt] = frag_a_f16(P + (size_t)(mt * 16) * PLD + kt * 32, PLD, lane);

#pragma unroll
    for (int nt = 0; nt < 8; ++nt) {
      v8f c = {};
#pragma unroll
      for (int kt = 0; kt < 12; ++kt) {
        const int n = lane & 15, ks = lane >> 4;
        int sc = s_base + kt * 32 + ks * 16;                 // chunk start in s
        sc = sc < 0 ? 0 : (sc > SS - 16 ? SS - 16 : sc);     // clamped region has P==0
        const _Float16* p = Vb + (size_t)(nt * 16 + n) * SS + sc;
        v8h lo = *(const v8h*)p;
        v8h hi = *(const v8h*)(p + 8);
        v16h bf = __builtin_shufflevector(lo, hi, 0,1,2,3,4,5,6,7,8,9,10,11,12,13,14,15);
        c = wmma16(a[kt], bf, c);
      }
      // AttnOut[b, s, h*D + dn], f16, heads concatenated
      _Float16* ob = AO + ((size_t)b * SS + win * 128 + mt * 16) * EE + h * DD + nt * 16;
      store_c_f16(ob, EE, lane, c);
    }
  }
}

// Output projection: Y[m, o] = sum_e AO[m,e] * Wo[o,e] + bo[o]; M = B*S, f32 out
// grid = M/16, block = 256 (8 waves, 8 n-tiles each)
__global__ __launch_bounds__(256) void out_proj(
    const _Float16* __restrict__ AO, const _Float16* __restrict__ Woh,
    const float* __restrict__ bo, float* __restrict__ out) {
  int mt = blockIdx.x;
  int wave = threadIdx.x >> 5, lane = threadIdx.x & 31;
  const _Float16* Ab = AO + (size_t)mt * 16 * EE;

  v8f c[8] = {};
  for (int kt = 0; kt < 32; ++kt) {
    v16h af = frag_a_f16(Ab + kt * 32, EE, lane);
    if (kt + 1 < 32) {
      // prefetch next A chunk (lowers to global_prefetch_b8)
      __builtin_prefetch(Ab + (kt + 1) * 32 + (size_t)(lane & 15) * EE, 0, 1);
    }
#pragma unroll
    for (int j = 0; j < 8; ++j) {
      int nt = wave * 8 + j;
      v16h bf = frag_b_f16(Woh + (size_t)(nt * 16) * EE + kt * 32, EE, lane);
      c[j] = wmma16(af, bf, c[j]);
    }
  }
  const int col = lane & 15, rh = lane >> 4;
#pragma unroll
  for (int j = 0; j < 8; ++j) {
    int nt = wave * 8 + j;
    float bias = bo[nt * 16 + col];
    float* ob = out + (size_t)mt * 16 * EE + nt * 16;
#pragma unroll
    for (int r = 0; r < 8; ++r) ob[(size_t)(r + rh * 8) * EE + col] = c[j][r] + bias;
  }
}

// ---------------- Host launcher ----------------
extern "C" void kernel_launch(void* const* d_in, const int* in_sizes, int n_in,
                              void* d_out, int out_size, void* d_ws, size_t ws_size,
                              hipStream_t stream) {
  (void)in_sizes; (void)n_in; (void)out_size; (void)ws_size;

  const float* values = (const float*)d_in[0];
  const float* keys   = (const float*)d_in[1];
  const float* query  = (const float*)d_in[2];
  const float* Wv     = (const float*)d_in[3];
  const float* Wk     = (const float*)d_in[4];
  const float* Wq     = (const float*)d_in[5];
  const float* Wo     = (const float*)d_in[6];
  const float* bo     = (const float*)d_in[7];

  char* ws = (char*)d_ws;
  const size_t WSMALL = (size_t)DD * DD;          // 16384
  const size_t WBIG   = (size_t)EE * EE;          // 1048576
  const size_t QKV    = (size_t)BB * SS * EE;     // 33554432

  _Float16* Wvh = (_Float16*)(ws);
  _Float16* Wkh = Wvh + WSMALL;
  _Float16* Wqh = Wkh + WSMALL;
  _Float16* Woh = Wqh + WSMALL;
  _Float16* Qh  = Woh + WBIG;
  _Float16* Kh  = Qh + QKV;
  _Float16* Vth = Kh + QKV;
  _Float16* AOh = Vth + QKV;

  // 1) weight conversion f32 -> f16
  cvt_f32_f16<<<(int)((WSMALL + 255) / 256), 256, 0, stream>>>(Wv, Wvh, (int)WSMALL);
  cvt_f32_f16<<<(int)((WSMALL + 255) / 256), 256, 0, stream>>>(Wk, Wkh, (int)WSMALL);
  cvt_f32_f16<<<(int)((WSMALL + 255) / 256), 256, 0, stream>>>(Wq, Wqh, (int)WSMALL);
  cvt_f32_f16<<<(int)((WBIG + 255) / 256), 256, 0, stream>>>(Wo, Woh, (int)WBIG);

  // 2) projections
  const int proj_grid = BB * HH * (SS / 64);  // 4096
  proj_rowmajor<<<proj_grid, 128, 0, stream>>>(query, Wqh, Qh);
  proj_rowmajor<<<proj_grid, 128, 0, stream>>>(keys,  Wkh, Kh);
  proj_v_t    <<<proj_grid, 128, 0, stream>>>(values, Wvh, Vth);

  // 3) windowed attention (299,008 B dynamic LDS; CDNA5 WGP has 320 KB)
  attn_window<<<BB * HH * NWIN, 128, ATTN_SMEM, stream>>>(Qh, Kh, Vth, AOh);

  // 4) output projection + bias -> f32 result
  out_proj<<<BB * SS / 16, 256, 0, stream>>>(AOh, Woh, bo, (float*)d_out);
}
